// HierarchicalPathReasoning_46866683134444
// MI455X (gfx1250) — matrix operand order repaired and checked by reading, not compiled, not run on hardware
//
#include <hip/hip_runtime.h>
#include <hip/hip_bf16.h>
#include <math.h>

// ---------------------------------------------------------------------------
// HierarchicalPathReasoning on MI455X (gfx1250, wave32, WMMA + TDM).
// ~160 GFLOP of dense matmul over ~5MB of L2-resident operands -> compute
// bound. All GEMMs run on v_wmma_f32_16x16x32_bf16 (fp32 accum). Weight
// K-slices are staged into LDS once per 4-wave block with the Tensor Data
// Mover (tensor_load_to_lds, TENSORcnt, double buffered, one barrier per
// K-step); B fragments are manually double-buffered in registers so DS
// latency overlaps the WMMA pipe.
// ---------------------------------------------------------------------------

typedef __attribute__((ext_vector_type(16))) __bf16 bf16x16;
typedef __attribute__((ext_vector_type(8)))  float  f32x8;
typedef unsigned int __attribute__((ext_vector_type(4))) u32x4;
typedef int          __attribute__((ext_vector_type(4))) i32x4;
typedef int          __attribute__((ext_vector_type(8))) i32x8;

#define N_NODES   1000
#define D_FEAT    256
#define P_PATHS   84000
#define TILE_P    16
#define WPB       4                              // 4 waves = 4 path tiles/block
#define N_TILES   (P_PATHS / TILE_P)             // 5250
#define NBLK_MAIN ((N_TILES + WPB - 1) / WPB)    // 1313

// LDS carve (dynamic shared): double-buffered TDM stage + per-wave buffers.
// Stage rows are padded to 80B (pad_interval=16DW, pad_amount=4DW) so the 16
// B-fragment rows of a tile land in distinct LDS banks.
#define STAGE_ROW_ELE 40                         // 32 data + 8 pad (bf16)
#define STAGE_BYTES   20480                      // 256 rows * 80B
#define OFF_STAGE     0
#define OFF_HBUF      (2 * STAGE_BYTES)          // 40960
#define OFF_FBUF      (OFF_HBUF + WPB * 8192)    // 73728
#define SMEM_BYTES    (OFF_FBUF + WPB * 24576)   // 172032

// ---- WMMA fragment helpers -------------------------------------------------
// A (16x32 bf16): lanes 0-15 row M=lane hold K=[0..15];
//                 lanes 16-31 same rows hold K=[16..31].
static __device__ inline bf16x16 frag_a(const __bf16* row, int hk) {
  union { bf16x16 v; uint4 q[2]; } u;
  u.q[0] = *(const uint4*)(row + 2 * hk);
  u.q[1] = *(const uint4*)(row + 2 * hk + 8);
  return u.v;
}
// B (32x16 bf16) from a staged LDS row (Wt[n][k], contiguous 32 K):
// lanes 0-15 (col n) hold K=[0..15]; lanes 16-31 hold K=[16..31].
static __device__ inline bf16x16 frag_bs(const __bf16* row, int h16) {
  union { bf16x16 v; uint4 q[2]; } u;
  const __bf16* p = row + h16;
  u.q[0] = *(const uint4*)(p);
  u.q[1] = *(const uint4*)(p + 8);
  return u.v;
}
static __device__ inline f32x8 wmma_bf16(bf16x16 a, bf16x16 b, f32x8 c) {
  return __builtin_amdgcn_wmma_f32_16x16x32_bf16(false, a, false, b, (short)0, c,
                                                 false, false);
}

// ---- TDM: stage a (nrows x 32) bf16 weight slice into LDS ------------------
// D# per CDNA5 ISA ch.8: group0 = {ctl, lds_addr, global_addr, type},
// group1 = {mask/size/pad, tensor dims, tile dims, strides}.
static __device__ inline void tdm_stage_slice(char* ldst, const __bf16* gsrc,
                                              int width_elems, int nrows) {
  unsigned long long ga = (unsigned long long)(size_t)gsrc;
  unsigned lds_off = (unsigned)(size_t)ldst;     // LDS aperture: low 32 = offset
  u32x4 g0 = {
    0x1u,                                        // count=1, user mode
    lds_off,
    (unsigned)ga,
    (unsigned)((ga >> 32) & 0x01FFFFFFu) | 0x80000000u  // addr[56:32] | type=2
  };
  unsigned d0 = (1u << 16)                       // data_size = 2B
              | (1u << 20)                       // pad_enable
              | (3u << 22)                       // pad_interval: 16 DWORDs
              | (3u << 25);                      // pad_amount:   4 DWORDs
  i32x8 g1 = {
    (int)d0,
    (int)((width_elems & 0xFFFF) << 16),                         // dim0 lo16
    (int)(((width_elems >> 16) & 0xFFFF) | ((nrows & 0xFFFF) << 16)),
    (int)(32u << 16),                                            // tile_dim0=32
    (int)(nrows & 0xFFFF),                                       // tile_dim1
    (int)width_elems,                                            // dim0_stride
    0, 0
  };
  i32x4 z4 = {0, 0, 0, 0};
#if __clang_major__ >= 23
  i32x8 z8 = {0, 0, 0, 0, 0, 0, 0, 0};
  __builtin_amdgcn_tensor_load_to_lds(g0, g1, z4, z4, z8, 0);
#else
  __builtin_amdgcn_tensor_load_to_lds(g0, g1, z4, z4, 0);
#endif
}

// ---- staged GEMM -----------------------------------------------------------
// One barrier per K-step: wave0 drains TENSORcnt (slice kk landed) before the
// barrier; after the barrier it launches the TDM for slice kk+1 into the idle
// buffer while all waves compute slice kk. B fragments are software-pipelined
// (prefetch nt+1 before WMMA nt) so ds_load latency hides under the WMMAs.
// A row for this lane comes from arowA (kk < splitKK) or arowB, contiguous K.
template <int NT>
static __device__ inline void gemm_staged(f32x8* acc, int ksteps, int splitKK,
                                          const __bf16* arowA, const __bf16* arowB,
                                          const __bf16* Bt, int bstride, int nrows,
                                          char* stage, int wave, int r, int hk,
                                          int h16) {
  if (wave == 0) tdm_stage_slice(stage, Bt, bstride, nrows);
  for (int kk = 0; kk < ksteps; ++kk) {
    if (wave == 0) __builtin_amdgcn_s_wait_tensorcnt(0);  // slice kk landed
    __syncthreads();   // kk visible to all; all done with the other buffer
    if (wave == 0 && kk + 1 < ksteps)
      tdm_stage_slice(stage + ((kk + 1) & 1) * STAGE_BYTES,
                      Bt + (kk + 1) * 32, bstride, nrows);   // overlaps compute
    const __bf16* buf = (const __bf16*)(stage + (kk & 1) * STAGE_BYTES);
    const __bf16* abase = (kk < splitKK) ? (arowA + kk * 32)
                                         : (arowB + (kk - splitKK) * 32);
    bf16x16 a = frag_a(abase, hk);
    bf16x16 b = frag_bs(buf + r * STAGE_ROW_ELE, h16);       // nt = 0
#pragma unroll
    for (int nt = 0; nt < NT; ++nt) {
      bf16x16 bn = b;
      if (nt + 1 < NT)
        bn = frag_bs(buf + ((nt + 1) * 16 + r) * STAGE_ROW_ELE, h16);
      acc[nt] = wmma_bf16(a, b, acc[nt]);
      b = bn;
    }
  }
}

// ---- prep kernels ----------------------------------------------------------
__global__ void cvt_bf16_kernel(const float* __restrict__ in,
                                __bf16* __restrict__ out, int n) {
  int i = blockIdx.x * blockDim.x + threadIdx.x;
  if (i < n) out[i] = (__bf16)in[i];
}

// (K, Nn) f32 row-major -> (Nn, K) bf16 row-major, batched over blockIdx.y
__global__ void transpose_bf16_kernel(const float* __restrict__ in,
                                      __bf16* __restrict__ out, int K, int Nn) {
  long long base = (long long)blockIdx.y * K * Nn;
  int idx = blockIdx.x * blockDim.x + threadIdx.x;
  if (idx < K * Nn) {
    int k = idx / Nn, n = idx % Nn;
    out[base + (long long)n * K + k] = (__bf16)in[base + idx];
  }
}

// ---- main pipeline ---------------------------------------------------------
__global__ __launch_bounds__(128) void path_pipeline(
    const __bf16* __restrict__ nf,          // (1000,256) bf16
    const int*    __restrict__ paths,       // (P,4)
    const int*    __restrict__ path_len,    // (P)
    const __bf16* __restrict__ W1t,         // (3,256,512)  [s][n][k]
    const float*  __restrict__ b1,
    const __bf16* __restrict__ W2t,         // (3,256,256)
    const float*  __restrict__ b2,
    const __bf16* __restrict__ Ws1t,        // (128,256)
    const float*  __restrict__ bs1,
    const float*  __restrict__ Ws2,         // (128)
    const float*  __restrict__ bs2,
    const __bf16* __restrict__ Wa1t,        // (256,1024)
    const float*  __restrict__ ba1,
    const __bf16* __restrict__ Wa2t,        // (256,256)
    const float*  __restrict__ ba2,
    float* __restrict__ partials,           // (NBLK_MAIN,256)
    float* __restrict__ scores)             // (P)
{
  extern __shared__ __align__(16) char smem[];
  const int lane  = threadIdx.x & 31;
  const int wave  = threadIdx.x >> 5;
  const int tile  = blockIdx.x * WPB + wave;
  const bool valid = tile < N_TILES;             // last block: 2 idle waves
  const int p0    = valid ? tile * TILE_P : 0;   // safe reads when invalid
  const int r     = lane & 15;                   // A row / B col / C col
  const int hk    = (lane >> 4) * 8;             // A K-half offset
  const int h16   = (lane >> 4) * 16;            // B K-half offset
  const int rbase = (lane >> 4) * 8;             // C row base for this lane

  char*   stage = smem + OFF_STAGE;
  __bf16* hbuf  = (__bf16*)(smem + OFF_HBUF + wave * 8192);
  __bf16* fbuf  = (__bf16*)(smem + OFF_FBUF + wave * 24576);
  float*  lpart = (float*)(smem + OFF_FBUF);     // aliases fbuf after last use

  const f32x8 fz = {0.f, 0.f, 0.f, 0.f, 0.f, 0.f, 0.f, 0.f};

  int pn[4];
#pragma unroll
  for (int j = 0; j < 4; ++j) pn[j] = paths[(p0 + r) * 4 + j];
  int pl[8];
#pragma unroll
  for (int q = 0; q < 8; ++q) pl[q] = path_len[p0 + rbase + q];

  // ===== per-step encoders: h = relu(cat@W1+b1); step = (h@W2+b2)*mask =====
  for (int s = 0; s < 3; ++s) {
    f32x8 acc[16];
#pragma unroll
    for (int nt = 0; nt < 16; ++nt) acc[nt] = fz;
    // A = [nf[cur] | nf[nxt]] gathered from global; B = W1t[s] via TDM stage
    gemm_staged<16>(acc, 16, 8, nf + pn[s] * D_FEAT, nf + pn[s + 1] * D_FEAT,
                    W1t + s * (256 * 512), 512, 256, stage, wave, r, hk, h16);
#pragma unroll
    for (int nt = 0; nt < 16; ++nt) {
      int col = nt * 16 + r;
      float bias = b1[s * 256 + col];
#pragma unroll
      for (int q = 0; q < 8; ++q) {
        float v = acc[nt][q] + bias;
        hbuf[(rbase + q) * 256 + col] = (__bf16)(v > 0.f ? v : 0.f);
      }
    }
    __syncthreads();

#pragma unroll
    for (int nt = 0; nt < 16; ++nt) acc[nt] = fz;
    gemm_staged<16>(acc, 8, 8, hbuf + r * 256, hbuf + r * 256,
                    W2t + s * (256 * 256), 256, 256, stage, wave, r, hk, h16);
#pragma unroll
    for (int nt = 0; nt < 16; ++nt) {
      int col = nt * 16 + r;
      float bias = b2[s * 256 + col];
#pragma unroll
      for (int q = 0; q < 8; ++q) {
        float m = (s + 1 < pl[q]) ? 1.f : 0.f;   // mask: s < path_len-1
        fbuf[(rbase + q) * 768 + s * 256 + col] = (__bf16)((acc[nt][q] + bias) * m);
      }
    }
    __syncthreads();
  }

  // ===== scores = sigmoid(relu(step0@Ws1+bs1)@Ws2+bs2) =====
  {
    f32x8 accS[8];
#pragma unroll
    for (int nt = 0; nt < 8; ++nt) accS[nt] = fz;
    gemm_staged<8>(accS, 8, 8, fbuf + r * 768, fbuf + r * 768,
                   Ws1t, 256, 128, stage, wave, r, hk, h16);
#pragma unroll
    for (int nt = 0; nt < 8; ++nt) {
      int col = nt * 16 + r;
      float bias = bs1[col];
#pragma unroll
      for (int q = 0; q < 8; ++q) {
        float v = accS[nt][q] + bias;
        hbuf[(rbase + q) * 128 + col] = (__bf16)(v > 0.f ? v : 0.f);
      }
    }
    __syncthreads();
    if (valid && lane < 16) {                    // 16x128 @ 128x1 matvec (VALU)
      float sum = 0.f;
      for (int j = 0; j < 128; ++j) sum += (float)hbuf[lane * 128 + j] * Ws2[j];
      float x = sum + bs2[0];
      scores[p0 + lane] = 1.f / (1.f + __expf(-x));
    }
    __syncthreads();
  }

  // ===== agg = relu(feat@Wa1+ba1)@Wa2+ba2 ; K=768 (padded block is zero) ====
  f32x8 acc[16];
#pragma unroll
  for (int nt = 0; nt < 16; ++nt) acc[nt] = fz;
  gemm_staged<16>(acc, 24, 24, fbuf + r * 768, fbuf + r * 768,
                  Wa1t, 1024, 256, stage, wave, r, hk, h16);
#pragma unroll
  for (int nt = 0; nt < 16; ++nt) {
    int col = nt * 16 + r;
    float bias = ba1[col];
#pragma unroll
    for (int q = 0; q < 8; ++q) {
      float v = acc[nt][q] + bias;
      hbuf[(rbase + q) * 256 + col] = (__bf16)(v > 0.f ? v : 0.f);
    }
  }
  __syncthreads();
#pragma unroll
  for (int nt = 0; nt < 16; ++nt) acc[nt] = fz;
  gemm_staged<16>(acc, 8, 8, hbuf + r * 256, hbuf + r * 256,
                  Wa2t, 256, 256, stage, wave, r, hk, h16);

  // column sums of (acc + ba2) over our 8 rows; zeroed for invalid tiles
  float colsum[16];
#pragma unroll
  for (int nt = 0; nt < 16; ++nt) {
    int col = nt * 16 + r;
    float s8 = 8.f * ba2[col];
#pragma unroll
    for (int q = 0; q < 8; ++q) s8 += acc[nt][q];
    colsum[nt] = valid ? s8 : 0.f;
  }
  __syncthreads();                               // fbuf dead -> lpart alias ok
  if (threadIdx.x < 128) {
    lpart[threadIdx.x * 2 + 0] = 0.f;
    lpart[threadIdx.x * 2 + 1] = 0.f;
  }
  __syncthreads();
#pragma unroll
  for (int nt = 0; nt < 16; ++nt)
    atomicAdd(&lpart[nt * 16 + r], colsum[nt]);  // ds_add_f32, block-local
  __syncthreads();
  if (threadIdx.x < 128) {
    partials[blockIdx.x * 256 + threadIdx.x * 2 + 0] = lpart[threadIdx.x * 2 + 0];
    partials[blockIdx.x * 256 + threadIdx.x * 2 + 1] = lpart[threadIdx.x * 2 + 1];
  }
}

// ---- deterministic reduction + broadcast add -------------------------------
__global__ void reduce_partials_kernel(const float* __restrict__ partials,
                                       float* __restrict__ agg, int nblocks) {
  int c = threadIdx.x;                           // 256 threads, 1 block
  float s = 0.f;
  for (int b = 0; b < nblocks; ++b) s += partials[b * 256 + c];
  agg[c] = s;
}

__global__ void final_add_kernel(const float* __restrict__ nodef,
                                 const float* __restrict__ agg,
                                 float* __restrict__ out) {
  int i = blockIdx.x * 256 + threadIdx.x;        // col = threadIdx.x (D=256)
  out[i] = nodef[i] + agg[threadIdx.x];
}

// ---------------------------------------------------------------------------
extern "C" void kernel_launch(void* const* d_in, const int* in_sizes, int n_in,
                              void* d_out, int out_size, void* d_ws, size_t ws_size,
                              hipStream_t stream) {
  (void)in_sizes; (void)n_in; (void)out_size; (void)ws_size;
  const float* nodef    = (const float*)d_in[0];
  const int*   paths    = (const int*)d_in[1];
  const int*   path_len = (const int*)d_in[2];
  const float* W1  = (const float*)d_in[3];
  const float* b1  = (const float*)d_in[4];
  const float* W2  = (const float*)d_in[5];
  const float* b2  = (const float*)d_in[6];
  const float* Ws1 = (const float*)d_in[7];
  const float* bs1 = (const float*)d_in[8];
  const float* Ws2 = (const float*)d_in[9];
  const float* bs2 = (const float*)d_in[10];
  const float* Wa1 = (const float*)d_in[11];
  const float* ba1 = (const float*)d_in[12];
  const float* Wa2 = (const float*)d_in[13];
  const float* ba2 = (const float*)d_in[14];

  char* ws = (char*)d_ws;
  size_t off = 0;
  auto alloc = [&](size_t bytes) {
    size_t o = off;
    off += (bytes + 4095) & ~(size_t)4095;
    return o;
  };
  __bf16* nf_b  = (__bf16*)(ws + alloc((size_t)N_NODES * D_FEAT * 2));
  __bf16* W1t   = (__bf16*)(ws + alloc((size_t)3 * 512 * 256 * 2));
  __bf16* W2t   = (__bf16*)(ws + alloc((size_t)3 * 256 * 256 * 2));
  __bf16* Wa1t  = (__bf16*)(ws + alloc((size_t)1024 * 256 * 2));
  __bf16* Wa2t  = (__bf16*)(ws + alloc((size_t)256 * 256 * 2));
  __bf16* Ws1t  = (__bf16*)(ws + alloc((size_t)256 * 128 * 2));
  float*  partials = (float*)(ws + alloc((size_t)NBLK_MAIN * 256 * 4));
  float*  agg   = (float*)(ws + alloc(256 * 4));

  float* out_updated = (float*)d_out;                   // (1000,256)
  float* out_scores  = out_updated + N_NODES * D_FEAT;  // (84000,)

  cvt_bf16_kernel<<<(N_NODES * D_FEAT + 255) / 256, 256, 0, stream>>>(
      nodef, nf_b, N_NODES * D_FEAT);
  transpose_bf16_kernel<<<dim3((512 * 256 + 255) / 256, 3), 256, 0, stream>>>(
      W1, W1t, 512, 256);
  transpose_bf16_kernel<<<dim3((256 * 256 + 255) / 256, 3), 256, 0, stream>>>(
      W2, W2t, 256, 256);
  transpose_bf16_kernel<<<dim3((1024 * 256 + 255) / 256, 1), 256, 0, stream>>>(
      Wa1, Wa1t, 1024, 256);
  transpose_bf16_kernel<<<dim3((256 * 256 + 255) / 256, 1), 256, 0, stream>>>(
      Wa2, Wa2t, 256, 256);
  transpose_bf16_kernel<<<dim3((256 * 128 + 255) / 256, 1), 256, 0, stream>>>(
      Ws1, Ws1t, 256, 128);

  hipFuncSetAttribute((const void*)path_pipeline,
                      hipFuncAttributeMaxDynamicSharedMemorySize, SMEM_BYTES);
  path_pipeline<<<NBLK_MAIN, 32 * WPB, SMEM_BYTES, stream>>>(
      nf_b, paths, path_len, W1t, b1, W2t, b2, Ws1t, bs1, Ws2, bs2, Wa1t, ba1,
      Wa2t, ba2, partials, out_scores);

  reduce_partials_kernel<<<1, 256, 0, stream>>>(partials, agg, NBLK_MAIN);
  final_add_kernel<<<N_NODES, 256, 0, stream>>>(nodef, agg, out_updated);
}
